// RoPEMultiHeadAttention_82755429859908
// MI455X (gfx1250) — compile-verified, matrix operand order checked by reading, not run on hardware
//
#include <hip/hip_runtime.h>
#include <hip/hip_bf16.h>
#include <math.h>

typedef _Float16 half_t;
typedef __attribute__((ext_vector_type(8)))  _Float16 v8h;
typedef __attribute__((ext_vector_type(16))) _Float16 v16h;
typedef __attribute__((ext_vector_type(8)))  float    v8f;
typedef __attribute__((ext_vector_type(4)))  int      i4v;

#define E_DIM 1024
#define S_LEN 2048
#define H_NUM 16
#define D_HEAD 64
#define M_ROWS 4096   // B*S

// ---------------------------------------------------------------- async copy (gfx1250 path w/ fallback)
#if defined(__AMDGCN__) && __has_builtin(__builtin_amdgcn_global_load_async_to_lds_b128)
#define USE_ASYNC 1
typedef __attribute__((address_space(1))) i4v g_i4v;  // global
typedef __attribute__((address_space(3))) i4v l_i4v;  // LDS
#else
#define USE_ASYNC 0
#endif

__device__ inline void cp_g2l(const half_t* g, half_t* l) {
#if USE_ASYNC
  __builtin_amdgcn_global_load_async_to_lds_b128(
      (g_i4v*)(void*)g, (l_i4v*)(void*)l, 0, 0);
#else
  *(uint4*)l = *(const uint4*)g;
#endif
}

__device__ inline void wait_async() {
#if USE_ASYNC
  asm volatile("s_wait_asynccnt 0" ::: "memory");
#endif
}

// ---------------------------------------------------------------- fragment helpers
__device__ inline v16h make_frag(const half_t* p0, const half_t* p1) {
  v8h lo = *(const v8h*)p0;
  v8h hi = *(const v8h*)p1;
  return __builtin_shufflevector(lo, hi, 0,1,2,3,4,5,6,7,8,9,10,11,12,13,14,15);
}

// A-frag (16xK row-major, lane = row) and B-frag (KxN with per-column
// k-contiguous storage, lane = col) share the same addressing pattern:
//   element i (0..7)  -> k = kbase + i ;  i (8..15) -> k = kbase + 16 + (i-8)
// where kbase = k0 + 8*(lane>>4).
__device__ inline v16h frag_ld(const half_t* base, int row, int ld, int kbase) {
  const half_t* p = base + row * ld + kbase;
  return make_frag(p, p + 16);
}

__device__ inline v8f wmma32(v16h a, v16h b, v8f c) {
  return __builtin_amdgcn_wmma_f32_16x16x32_f16(false, a, false, b,
                                                (short)0, c, false, false);
}

// ---------------------------------------------------------------- cast
__global__ __launch_bounds__(256) void cast_f32_f16(const float* __restrict__ src,
                                                    half_t* __restrict__ dst, int n8) {
  int i = blockIdx.x * blockDim.x + threadIdx.x;
  if (i >= n8) return;
  const float4* s = (const float4*)src + 2 * i;
  float4 a = s[0], b = s[1];
  v8h o;
  o[0] = (half_t)a.x; o[1] = (half_t)a.y; o[2] = (half_t)a.z; o[3] = (half_t)a.w;
  o[4] = (half_t)b.x; o[5] = (half_t)b.y; o[6] = (half_t)b.z; o[7] = (half_t)b.w;
  *((v8h*)dst + i) = o;
}

// ---------------------------------------------------------------- GEMM
// C[M=4096, N=1024] = A[4096x1024] x Bw^T  (+bias), A,Bw f16, Bw row-major [N][K]
// mode 0: write f16 head-major [B,H,S,D];  mode 1: write f32 row-major [M,N]
// Double-buffered LDS pipeline: async-copy tile i+1 while WMMAs consume tile i.
__global__ __launch_bounds__(128) void gemm_wmma(const half_t* __restrict__ A,
                                                 const half_t* __restrict__ Bw,
                                                 const float* __restrict__ bias,
                                                 half_t* __restrict__ out_h,
                                                 float* __restrict__ out_f,
                                                 int mode) {
  __shared__ __align__(16) half_t As[2][64 * 64];
  __shared__ __align__(16) half_t Bs[2][64 * 64];

  const int bn0 = blockIdx.x * 64;
  const int bm0 = blockIdx.y * 64;
  const int t = threadIdx.x;
  const int lane = t & 31, w = t >> 5;
  const int ln = lane & 15, hf = lane >> 4;
  const int srow = t >> 3;          // 0..15
  const int soff = (t & 7) << 3;    // 0,8,...,56 halves (16B chunks)

  v8f acc[4];
#pragma unroll
  for (int j = 0; j < 4; ++j) acc[j] = (v8f){0,0,0,0,0,0,0,0};

  auto stage = [&](int buf, int kb) {
#pragma unroll
    for (int it = 0; it < 4; ++it) {
      int r2 = srow + it * 16;
      cp_g2l(&A[(size_t)(bm0 + r2) * E_DIM + kb + soff], &As[buf][r2 * 64 + soff]);
      cp_g2l(&Bw[(size_t)(bn0 + r2) * E_DIM + kb + soff], &Bs[buf][r2 * 64 + soff]);
    }
  };

  stage(0, 0);
  int ibuf = 0;
  for (int kb = 0; kb < E_DIM; kb += 64, ibuf ^= 1) {
    wait_async();
    __syncthreads();
    if (kb + 64 < E_DIM) stage(ibuf ^ 1, kb + 64);

    const half_t* curA = As[ibuf];
    const half_t* curB = Bs[ibuf];
#pragma unroll
    for (int kk = 0; kk < 2; ++kk) {
      int kbase = kk * 32 + 8 * hf;
      v16h af = frag_ld(curA, 16 * w + ln, 64, kbase);
#pragma unroll
      for (int j = 0; j < 4; ++j) {
        v16h bf = frag_ld(curB, 16 * j + ln, 64, kbase);
        acc[j] = wmma32(af, bf, acc[j]);
      }
    }
    __syncthreads();
  }

#pragma unroll
  for (int j = 0; j < 4; ++j) {
    int col = bn0 + 16 * j + ln;
    float bv = bias[col];
#pragma unroll
    for (int r = 0; r < 8; ++r) {
      int row = bm0 + 16 * w + r + 8 * hf;
      float val = acc[j][r] + bv;
      if (mode == 0) {
        int b = row >> 11, s = row & 2047;
        int h = col >> 6, d = col & 63;
        out_h[(((size_t)(b * H_NUM + h)) * S_LEN + s) * D_HEAD + d] = (half_t)val;
      } else {
        out_f[(size_t)row * E_DIM + col] = val;
      }
    }
  }
}

// ---------------------------------------------------------------- RoPE (in-place on Q,K f16 head-major)
__global__ __launch_bounds__(256) void rope_kernel(half_t* __restrict__ Q,
                                                   half_t* __restrict__ K) {
  int idx = blockIdx.x * blockDim.x + threadIdx.x;   // 2 * 32 * 2048 * 32
  int tsel = idx >> 21;
  int rem  = idx & ((1 << 21) - 1);
  int d  = rem & 31;
  int s  = (rem >> 5) & 2047;
  int bh = rem >> 16;
  half_t* p = (tsel ? K : Q) + ((size_t)bh * S_LEN + s) * D_HEAD;
  float f   = expf(-(float)d * 0.28782313662425572f);  // ln(10000)/32
  float sn, cs;
  sincosf((float)s * f, &sn, &cs);
  float x1 = (float)p[d], x2 = (float)p[d + 32];
  p[d]      = (half_t)(x1 * cs - x2 * sn);
  p[d + 32] = (half_t)(x2 * cs + x1 * sn);
}

// ---------------------------------------------------------------- Flash attention (causal)
// grid = (S/64, B*H); block = 128 (4 waves). Wave w owns query rows qb0+16w..+15.
__global__ __launch_bounds__(128) void attn_wmma(const half_t* __restrict__ Q,
                                                 const half_t* __restrict__ Kx,
                                                 const half_t* __restrict__ V,
                                                 half_t* __restrict__ Oout) {
  __shared__ __align__(16) half_t Ks[64 * 64];     // [key][d]
  __shared__ __align__(16) half_t Vts[64 * 64];    // [d][key]
  __shared__ __align__(16) half_t Ps[4][16 * 64];  // per-wave P tile [row][key]

  const int bh = blockIdx.y;
  const int qb0 = blockIdx.x * 64;
  const int t = threadIdx.x;
  const int lane = t & 31, w = t >> 5;
  const int ln = lane & 15, hf = lane >> 4;
  const int qrowbase = qb0 + 16 * w;
  const int srow = t >> 3;
  const int soff = (t & 7) << 3;

  const half_t* Qb = Q  + (size_t)bh * S_LEN * D_HEAD;
  const half_t* Kb = Kx + (size_t)bh * S_LEN * D_HEAD;
  const half_t* Vb = V  + (size_t)bh * S_LEN * D_HEAD;

  // Q fragments for this wave's 16 rows (K-dim 64 -> 2 frags), kept in registers.
  v16h qf[2];
#pragma unroll
  for (int kk = 0; kk < 2; ++kk)
    qf[kk] = frag_ld(Qb, qrowbase + ln, D_HEAD, kk * 32 + 8 * hf);

  float m_[8], lp[8];
  v8f o[4];
#pragma unroll
  for (int r = 0; r < 8; ++r) { m_[r] = -1e30f; lp[r] = 0.f; }
#pragma unroll
  for (int j = 0; j < 4; ++j) o[j] = (v8f){0,0,0,0,0,0,0,0};

  for (int kb0 = 0; kb0 <= qb0; kb0 += 64) {
    __syncthreads();
#pragma unroll
    for (int it = 0; it < 4; ++it) {
      int r2 = srow + it * 16;
      cp_g2l(&Kb[(size_t)(kb0 + r2) * D_HEAD + soff], &Ks[r2 * 64 + soff]);
      v8h vv = *(const v8h*)&Vb[(size_t)(kb0 + r2) * D_HEAD + soff];
#pragma unroll
      for (int u = 0; u < 8; ++u) Vts[(soff + u) * 64 + r2] = vv[u];
    }
    wait_async();
    __syncthreads();

    // S = Q * K^T  (16 x 64)
    v8f sf[4];
#pragma unroll
    for (int j = 0; j < 4; ++j) sf[j] = (v8f){0,0,0,0,0,0,0,0};
#pragma unroll
    for (int kk = 0; kk < 2; ++kk) {
      int kbase = kk * 32 + 8 * hf;
#pragma unroll
      for (int j = 0; j < 4; ++j) {
        v16h bf = frag_ld(Ks, 16 * j + ln, 64, kbase);
        sf[j] = wmma32(qf[kk], bf, sf[j]);
      }
    }

    const bool diag = (kb0 == qb0);
#pragma unroll
    for (int j = 0; j < 4; ++j) {
      int keyc = kb0 + 16 * j + ln;
#pragma unroll
      for (int r = 0; r < 8; ++r) {
        float sv = sf[j][r] * 0.125f;   // 1/sqrt(64)
        if (diag && keyc > qrowbase + r + 8 * hf) sv = -1e30f;
        sf[j][r] = sv;
      }
    }

    // online softmax: row stats reduce across lane bits 0..3 (lanes sharing a row group)
    float rm[8];
#pragma unroll
    for (int r = 0; r < 8; ++r) {
      float v0 = fmaxf(fmaxf(sf[0][r], sf[1][r]), fmaxf(sf[2][r], sf[3][r]));
      v0 = fmaxf(v0, __shfl_xor(v0, 1, 32));
      v0 = fmaxf(v0, __shfl_xor(v0, 2, 32));
      v0 = fmaxf(v0, __shfl_xor(v0, 4, 32));
      v0 = fmaxf(v0, __shfl_xor(v0, 8, 32));
      rm[r] = v0;
    }
    float alpha[8], ps[8];
#pragma unroll
    for (int r = 0; r < 8; ++r) {
      float mn = fmaxf(m_[r], rm[r]);
      alpha[r] = __expf(m_[r] - mn);
      m_[r] = mn;
      ps[r] = 0.f;
    }
#pragma unroll
    for (int j = 0; j < 4; ++j) {
#pragma unroll
      for (int r = 0; r < 8; ++r) {
        float p = __expf(sf[j][r] - m_[r]);
        ps[r] += p;
        Ps[w][(r + 8 * hf) * 64 + 16 * j + ln] = (half_t)p;
      }
    }
#pragma unroll
    for (int r = 0; r < 8; ++r) lp[r] = lp[r] * alpha[r] + ps[r];
#pragma unroll
    for (int j = 0; j < 4; ++j)
#pragma unroll
      for (int r = 0; r < 8; ++r) o[j][r] *= alpha[r];

    __builtin_amdgcn_wave_barrier();   // keep P stores before P frag loads (same-wave LDS is in-order)

    // O += P * V   (P: 16 x 64keys as A;  V: 64keys x 64d as B via Vts[d][key])
#pragma unroll
    for (int kk = 0; kk < 2; ++kk) {
      int kbase = kk * 32 + 8 * hf;
      v16h pf = frag_ld(&Ps[w][0], ln, 64, kbase);
#pragma unroll
      for (int j = 0; j < 4; ++j) {
        v16h vf = frag_ld(Vts, 16 * j + ln, 64, kbase);
        o[j] = wmma32(pf, vf, o[j]);
      }
    }
  }

  float lt[8];
#pragma unroll
  for (int r = 0; r < 8; ++r) {
    float v0 = lp[r];
    v0 += __shfl_xor(v0, 1, 32);
    v0 += __shfl_xor(v0, 2, 32);
    v0 += __shfl_xor(v0, 4, 32);
    v0 += __shfl_xor(v0, 8, 32);
    lt[r] = v0;
  }

  const int b = bh >> 4, h = bh & 15;
#pragma unroll
  for (int j = 0; j < 4; ++j) {
    int e = h * D_HEAD + 16 * j + ln;
#pragma unroll
    for (int r = 0; r < 8; ++r) {
      int sr = qrowbase + r + 8 * hf;
      size_t n = (size_t)b * S_LEN + sr;
      Oout[n * E_DIM + e] = (half_t)(o[j][r] / lt[r]);
    }
  }
}

// ---------------------------------------------------------------- launch
extern "C" void kernel_launch(void* const* d_in, const int* in_sizes, int n_in,
                              void* d_out, int out_size, void* d_ws, size_t ws_size,
                              hipStream_t stream) {
  const float* q_in = (const float*)d_in[0];
  const float* k_in = (const float*)d_in[1];
  const float* v_in = (const float*)d_in[2];
  const float* Wq = (const float*)d_in[3];  const float* bq = (const float*)d_in[4];
  const float* Wk = (const float*)d_in[5];  const float* bk = (const float*)d_in[6];
  const float* Wv = (const float*)d_in[7];  const float* bv = (const float*)d_in[8];
  const float* Wo = (const float*)d_in[9];  const float* bo = (const float*)d_in[10];
  float* out = (float*)d_out;

  char* ws = (char*)d_ws;
  const size_t MB = 1u << 20;
  half_t* Xq16 = (half_t*)(ws + 0 * MB);
  half_t* Xk16 = (half_t*)(ws + 8 * MB);
  half_t* Xv16 = (half_t*)(ws + 16 * MB);
  half_t* Wq16 = (half_t*)(ws + 24 * MB);
  half_t* Wk16 = (half_t*)(ws + 26 * MB);
  half_t* Wv16 = (half_t*)(ws + 28 * MB);
  half_t* Wo16 = (half_t*)(ws + 30 * MB);
  half_t* Q16  = (half_t*)(ws + 32 * MB);
  half_t* K16  = (half_t*)(ws + 40 * MB);
  half_t* V16  = (half_t*)(ws + 48 * MB);
  half_t* AO16 = (half_t*)(ws + 56 * MB);

  const int nAct = M_ROWS * E_DIM;       // 4,194,304
  const int nW   = E_DIM * E_DIM;        // 1,048,576
  auto cast = [&](const float* s, half_t* d, int n) {
    int n8 = n / 8;
    cast_f32_f16<<<(n8 + 255) / 256, 256, 0, stream>>>(s, d, n8);
  };
  cast(q_in, Xq16, nAct);
  cast(k_in, Xk16, nAct);
  cast(v_in, Xv16, nAct);
  cast(Wq, Wq16, nW);
  cast(Wk, Wk16, nW);
  cast(Wv, Wv16, nW);
  cast(Wo, Wo16, nW);

  dim3 ggrid(E_DIM / 64, M_ROWS / 64);
  gemm_wmma<<<ggrid, 128, 0, stream>>>(Xq16, Wq16, bq, Q16, nullptr, 0);
  gemm_wmma<<<ggrid, 128, 0, stream>>>(Xk16, Wk16, bk, K16, nullptr, 0);
  gemm_wmma<<<ggrid, 128, 0, stream>>>(Xv16, Wv16, bv, V16, nullptr, 0);

  int ropeN = 2 * 32 * S_LEN * 32;       // both tensors, all rotation pairs
  rope_kernel<<<ropeN / 256, 256, 0, stream>>>(Q16, K16);

  attn_wmma<<<dim3(S_LEN / 64, 32), 128, 0, stream>>>(Q16, K16, V16, AO16);

  gemm_wmma<<<ggrid, 128, 0, stream>>>(AO16, Wo16, bo, nullptr, out, 1);
}
